// LinearTransformerAgent_41747082117809
// MI455X (gfx1250) — matrix operand order repaired and checked by reading, not compiled, not run on hardware
//
#include <hip/hip_runtime.h>
#include <hip/hip_bf16.h>
#include <math.h>
#include <stdint.h>

// ---------------------------------------------------------------------------
// Problem constants (from reference)
// ---------------------------------------------------------------------------
#define BB 64
#define TT 512
#define DOBS 64
#define DD 512
#define HH 8
#define DH 64
#define LL 4
#define NACTS 16
#define DFF 2048
#define ROWS (BB * TT)          // 32768

typedef __attribute__((ext_vector_type(16))) _Float16 v16h;
typedef __attribute__((ext_vector_type(8)))  _Float16 v8h;
typedef __attribute__((ext_vector_type(8)))  float    v8f;

// ---------------------------------------------------------------------------
// WMMA helpers (gfx1250: v_wmma_f32_16x16x32_f16, wave32)
// A fragment: 16x32 (MxK) row-major source, lda in halves.
// B fragment: 32x16 (KxN); source stored N-major ("transposed", ldb in halves)
// ---------------------------------------------------------------------------
static __device__ __forceinline__ v16h load_a_frag(const _Float16* A, int lda,
                                                   int m0, int k0, int lane) {
  int r  = lane & 15;
  int kb = (lane >> 4) << 3;                  // 0 or 8
  const _Float16* p = A + (size_t)(m0 + r) * lda + k0 + kb;
  v16h a;
#pragma unroll
  for (int i = 0; i < 8; ++i) { a[i] = p[i]; a[i + 8] = p[i + 16]; }
  return a;
}

static __device__ __forceinline__ v16h load_b_frag(const _Float16* Bt, int ldb,
                                                   int n0, int k0, int lane) {
  int c  = lane & 15;
  int kb = (lane >> 4) << 3;                  // 0 or 8
  const _Float16* p = Bt + (size_t)(n0 + c) * ldb + k0 + kb;
  v16h b;
#pragma unroll
  for (int i = 0; i < 8; ++i) { b[i] = p[i]; b[i + 8] = p[i + 16]; }
  return b;
}

static __device__ __forceinline__ v8f wmma_f16(v16h a, v16h b, v8f c) {
  return __builtin_amdgcn_wmma_f32_16x16x32_f16(false, a, false, b,
                                                (short)0, c, false, false);
}

static __device__ __forceinline__ float gelu_tanh(float v) {
  const float k = 0.7978845608028654f;        // sqrt(2/pi)
  float u = k * (v + 0.044715f * v * v * v);
  return 0.5f * v * (1.0f + tanhf(u));
}

// ---------------------------------------------------------------------------
// CDNA5 async global->LDS DMA (ASYNCcnt-tracked).
//   lds_off: byte offset of destination in LDS (AS3 offset = low 32 bits of
//            a generic pointer to a __shared__ object on AMDGPU)
//   gptr:    global source address (16 bytes copied per active lane)
// ---------------------------------------------------------------------------
static __device__ __forceinline__ void async_copy_b128(uint32_t lds_off,
                                                       const void* gptr) {
  asm volatile("global_load_async_to_lds_b128 %0, %1, off"
               :: "v"(lds_off), "v"(gptr) : "memory");
}
static __device__ __forceinline__ void wait_async0() {
  asm volatile("s_wait_asynccnt 0x0" ::: "memory");
}

// ---------------------------------------------------------------------------
// Weight convert + transpose: src[K][N] f32 -> dst[N][K] f16
// ---------------------------------------------------------------------------
__global__ void cvt_t_kernel(const float* __restrict__ src,
                             _Float16* __restrict__ dst, int K, int N) {
  size_t total = (size_t)K * N;
  for (size_t idx = (size_t)blockIdx.x * blockDim.x + threadIdx.x; idx < total;
       idx += (size_t)gridDim.x * blockDim.x) {
    size_t k = idx / N, n = idx % N;
    dst[n * (size_t)K + k] = (_Float16)src[idx];
  }
}

// ---------------------------------------------------------------------------
// Embedding: x = obs@W_obs + b_obs + E_act[a] + r*W_rew + b_rew + E_time[t]
// ---------------------------------------------------------------------------
__global__ void embed_kernel(const float* __restrict__ obs,
                             const int* __restrict__ act,
                             const float* __restrict__ rew,
                             const int* __restrict__ sobs,
                             const float* __restrict__ W_obs,
                             const float* __restrict__ b_obs,
                             const float* __restrict__ E_act,
                             const float* __restrict__ W_rew,
                             const float* __restrict__ b_rew,
                             const float* __restrict__ E_time,
                             float* __restrict__ xout) {
  int row = blockIdx.x;                  // b*T + t
  int b = row >> 9, t = row & (TT - 1);
  int a = act[row];
  float r = rew[row];
  int tim = sobs[b] + t;
  const float* orow = obs + (size_t)row * DOBS;
  for (int d = threadIdx.x; d < DD; d += blockDim.x) {
    float acc = b_obs[d] + E_act[(size_t)a * DD + d] + r * W_rew[d] + b_rew[d] +
                E_time[(size_t)tim * DD + d];
#pragma unroll 8
    for (int k = 0; k < DOBS; ++k) acc += orow[k] * W_obs[(size_t)k * DD + d];
    xout[(size_t)row * DD + d] = acc;
  }
}

// ---------------------------------------------------------------------------
// LayerNorm over D=512, f32 in -> f16 out. One wave per row, 8 waves/block.
// ---------------------------------------------------------------------------
__global__ void ln_kernel(const float* __restrict__ x, _Float16* __restrict__ out,
                          const float* __restrict__ s, const float* __restrict__ b) {
  int lane = threadIdx.x & 31;
  int row = blockIdx.x * (blockDim.x >> 5) + (threadIdx.x >> 5);
  const float* xr = x + (size_t)row * DD;
  float sum = 0.f, sq = 0.f;
#pragma unroll
  for (int i = lane; i < DD; i += 32) { float v = xr[i]; sum += v; sq += v * v; }
#pragma unroll
  for (int off = 16; off > 0; off >>= 1) {
    sum += __shfl_xor(sum, off, 32);
    sq  += __shfl_xor(sq,  off, 32);
  }
  float mean = sum * (1.0f / DD);
  float rstd = rsqrtf(sq * (1.0f / DD) - mean * mean + 1e-6f);
  _Float16* orow = out + (size_t)row * DD;
#pragma unroll
  for (int i = lane; i < DD; i += 32)
    orow[i] = (_Float16)((xr[i] - mean) * rstd * s[i] + b[i]);
}

// ---------------------------------------------------------------------------
// Generic WMMA GEMM: C[M,N] = A[M,K](f16) x Wt[N,K](f16) + bias
//   MODE 0: store f16 | MODE 1: gelu+store f16 | MODE 2: += into f32 (residual)
// block = 256 (8 waves, 2x4); block tile 64x128; 32x32 per wave with 4
// independent accumulators (1 fragment load per WMMA, hazard-free interleave).
// grid = (N/128, M/64)
// ---------------------------------------------------------------------------
template <int MODE>
static __device__ __forceinline__ void store_tile(v8f c, int m0, int n0, int lane,
                                                  const float* __restrict__ bias,
                                                  _Float16* __restrict__ outh,
                                                  float* __restrict__ outf, int N) {
  int col = n0 + (lane & 15);
  float bn = bias[col];
  int rbase = m0 + ((lane >> 4) << 3);
#pragma unroll
  for (int r = 0; r < 8; ++r) {
    int row = rbase + r;
    float v = c[r] + bn;
    if (MODE == 1) v = gelu_tanh(v);
    if (MODE == 2) outf[(size_t)row * N + col] += v;
    else           outh[(size_t)row * N + col] = (_Float16)v;
  }
}

template <int MODE>
__global__ void gemm_kernel(const _Float16* __restrict__ A,
                            const _Float16* __restrict__ Bt,
                            const float* __restrict__ bias,
                            _Float16* __restrict__ outh,
                            float* __restrict__ outf, int N, int K) {
  int lane = threadIdx.x & 31;
  int wave = threadIdx.x >> 5;
  int mw = wave >> 2;                     // 0..1
  int nw = wave & 3;                      // 0..3
  int m0 = blockIdx.y * 64 + mw * 32;
  int n0 = blockIdx.x * 128 + nw * 32;
  v8f c00 = {}, c01 = {}, c10 = {}, c11 = {};
#pragma unroll 2
  for (int k0 = 0; k0 < K; k0 += 32) {
    // prefetch next weight slab (global_prefetch_b8)
    __builtin_prefetch(Bt + (size_t)(n0 + (lane & 15)) * K + k0 + 64, 0, 0);
    v16h a0 = load_a_frag(A, K, m0,      k0, lane);
    v16h a1 = load_a_frag(A, K, m0 + 16, k0, lane);
    v16h b0 = load_b_frag(Bt, K, n0,      k0, lane);
    v16h b1 = load_b_frag(Bt, K, n0 + 16, k0, lane);
    c00 = wmma_f16(a0, b0, c00);
    c01 = wmma_f16(a0, b1, c01);
    c10 = wmma_f16(a1, b0, c10);
    c11 = wmma_f16(a1, b1, c11);
  }
  store_tile<MODE>(c00, m0,      n0,      lane, bias, outh, outf, N);
  store_tile<MODE>(c01, m0,      n0 + 16, lane, bias, outh, outf, N);
  store_tile<MODE>(c10, m0 + 16, n0,      lane, bias, outh, outf, N);
  store_tile<MODE>(c11, m0 + 16, n0 + 16, lane, bias, outh, outf, N);
}

// ---------------------------------------------------------------------------
// Chunked causal linear attention, one block (128 thr = 4 waves) per (h, b).
// qkv: [B][T][3*D] f16 (n = i*512 + h*64 + dh). Chunk = 32 tokens, 16 chunks.
//   out_c = tril(q_c k_c^T) v_c + q_c S ;  S += k_c^T v_c
// Chunk staging is double-buffered async DMA (global_load_async_to_lds_b128):
// chunk c+1 streams into LDS while chunk c's WMMAs run.
// Wave w owns S/out columns e in [w*16, w*16+16).
// ---------------------------------------------------------------------------
__global__ void attn_kernel(const _Float16* __restrict__ qkv,
                            const float* __restrict__ s_in,
                            float* __restrict__ s_out,
                            _Float16* __restrict__ obuf, int layer) {
  int h = blockIdx.x, b = blockIdx.y;
  int lane = threadIdx.x & 31, w = threadIdx.x >> 5;

  __shared__ _Float16 sQ[2][32 * 64];   // token-major q chunk (async dst)
  __shared__ _Float16 sK[2][32 * 64];   // token-major k chunk (async dst)
  __shared__ _Float16 sV[2][32 * 64];   // token-major v chunk (async dst)
  __shared__ _Float16 sKt[64 * 32];     // dh-major k chunk (A for S update)
  __shared__ _Float16 sVt[64 * 32];     // dh-major v chunk (B operands)
  __shared__ _Float16 sSt[64 * 64];     // sSt[e*64+d] = S[d][e] (B for q@S)
  __shared__ _Float16 sP[32 * 32];      // masked scores (A for P@V)

  const float* sb = s_in + (((size_t)b * LL + layer) * HH + h) * (DH * DH);
  int e   = w * 16 + (lane & 15);
  int dhi = (lane >> 4) << 3;

  v8f acc[4];
#pragma unroll
  for (int mt = 0; mt < 4; ++mt)
#pragma unroll
    for (int r = 0; r < 8; ++r)
      acc[mt][r] = sb[(size_t)(mt * 16 + dhi + r) * DH + e];
  for (int idx = threadIdx.x; idx < DH * DH; idx += 128) {
    int d = idx & 63, ee = idx >> 6;
    sSt[ee * 64 + d] = (_Float16)sb[(size_t)d * DH + ee];
  }

  const _Float16* base = qkv + ((size_t)b * TT) * (3 * DD) + h * DH;

  // issue async DMA of one 32x64 q/k/v chunk into buffer `buf`
  // (256 16-byte units per matrix; 128 threads x 2 units; 6 asyncs per thread)
  auto issue_chunk = [&](int c, int buf) {
#pragma unroll
    for (int u0 = 0; u0 < 2; ++u0) {
      int u = threadIdx.x + u0 * 128;             // 0..255
      int tok = u >> 3, seg = (u & 7) << 3;       // seg: halves 0,8,...,56
      const _Float16* gq = base + (size_t)(c * 32 + tok) * (3 * DD) + seg;
      uint32_t loff = (uint32_t)(uintptr_t)&sQ[buf][tok * 64 + seg];
      async_copy_b128(loff, gq);
      loff = (uint32_t)(uintptr_t)&sK[buf][tok * 64 + seg];
      async_copy_b128(loff, gq + DD);
      loff = (uint32_t)(uintptr_t)&sV[buf][tok * 64 + seg];
      async_copy_b128(loff, gq + 2 * DD);
    }
  };

  issue_chunk(0, 0);

  for (int c = 0; c < TT / 32; ++c) {
    int t0 = c * 32;
    int cur = c & 1;
    // wait for this chunk's DMA, make it visible to all waves
    wait_async0();
    __syncthreads();
    // overlap: stream next chunk into the other buffer during compute
    if (c + 1 < TT / 32) issue_chunk(c + 1, cur ^ 1);

    // ---- build dh-major transposed copies of k/v (LDS -> LDS) ----
    for (int idx = threadIdx.x; idx < 32 * 64; idx += 128) {
      int tok = idx >> 6, d = idx & 63;
      sKt[d * 32 + tok] = sK[cur][idx];
      sVt[d * 32 + tok] = sV[cur][idx];
    }
    __syncthreads();

    // ---- scores = tril(q k^T), wave w -> 16x16 tile (i=w>>1, j=w&1) ----
    {
      int ti = (w >> 1) * 16, tj = (w & 1) * 16;
      v8f sc = {};
      sc = wmma_f16(load_a_frag(sQ[cur], 64, ti, 0,  lane),
                    load_b_frag(sK[cur], 64, tj, 0,  lane), sc);
      sc = wmma_f16(load_a_frag(sQ[cur], 64, ti, 32, lane),
                    load_b_frag(sK[cur], 64, tj, 32, lane), sc);
      int tc = tj + (lane & 15);
      int trb = ti + dhi;
#pragma unroll
      for (int r = 0; r < 8; ++r) {
        int tr = trb + r;
        sP[tr * 32 + tc] = (_Float16)((tr >= tc) ? sc[r] : 0.0f);
      }
    }
    __syncthreads();

    // ---- out = P @ V + Q @ S  (wave w -> dh cols e; 2 indep accumulators) ----
    {
      v8f o0 = {}, o1 = {};
      o0 = wmma_f16(load_a_frag(sP, 32, 0,  0,  lane), load_b_frag(sVt, 32, w * 16, 0,  lane), o0);
      o1 = wmma_f16(load_a_frag(sP, 32, 16, 0,  lane), load_b_frag(sVt, 32, w * 16, 0,  lane), o1);
      o0 = wmma_f16(load_a_frag(sQ[cur], 64, 0,  0,  lane), load_b_frag(sSt, 64, w * 16, 0,  lane), o0);
      o1 = wmma_f16(load_a_frag(sQ[cur], 64, 16, 0,  lane), load_b_frag(sSt, 64, w * 16, 0,  lane), o1);
      o0 = wmma_f16(load_a_frag(sQ[cur], 64, 0,  32, lane), load_b_frag(sSt, 64, w * 16, 32, lane), o0);
      o1 = wmma_f16(load_a_frag(sQ[cur], 64, 16, 32, lane), load_b_frag(sSt, 64, w * 16, 32, lane), o1);
      int col = h * DH + e;
#pragma unroll
      for (int r = 0; r < 8; ++r) {
        int t0r = t0 + dhi + r;
        obuf[((size_t)b * TT + t0r) * DD + col]      = (_Float16)o0[r];
        obuf[((size_t)b * TT + t0r + 16) * DD + col] = (_Float16)o1[r];
      }
    }
    __syncthreads();

    // ---- S += K^T V (wave w -> e cols; 4 independent m-tiles) + refresh sSt ----
#pragma unroll
    for (int mt = 0; mt < 4; ++mt)
      acc[mt] = wmma_f16(load_a_frag(sKt, 32, mt * 16, 0, lane),
                         load_b_frag(sVt, 32, w * 16, 0, lane), acc[mt]);
#pragma unroll
    for (int mt = 0; mt < 4; ++mt)
#pragma unroll
      for (int r = 0; r < 8; ++r)
        sSt[e * 64 + (mt * 16 + dhi + r)] = (_Float16)acc[mt][r];
    __syncthreads();
  }

  float* so = s_out + (((size_t)b * LL + layer) * HH + h) * (DH * DH);
#pragma unroll
  for (int mt = 0; mt < 4; ++mt)
#pragma unroll
    for (int r = 0; r < 8; ++r)
      so[(size_t)(mt * 16 + dhi + r) * DH + e] = acc[mt][r];
}

// ---------------------------------------------------------------------------
// Final LN + actor/critic heads. One wave per row.
// ---------------------------------------------------------------------------
__global__ void head_kernel(const float* __restrict__ x,
                            const float* __restrict__ lnf_s,
                            const float* __restrict__ lnf_b,
                            const float* __restrict__ W_actor,
                            const float* __restrict__ b_actor,
                            const float* __restrict__ W_critic,
                            const float* __restrict__ b_critic,
                            float* __restrict__ logits,
                            float* __restrict__ val) {
  __shared__ float xr[DD];
  int row = blockIdx.x;
  int lane = threadIdx.x;
  const float* xp = x + (size_t)row * DD;
  float sum = 0.f, sq = 0.f;
#pragma unroll
  for (int i = lane; i < DD; i += 32) { float v = xp[i]; sum += v; sq += v * v; }
#pragma unroll
  for (int off = 16; off > 0; off >>= 1) {
    sum += __shfl_xor(sum, off, 32);
    sq  += __shfl_xor(sq,  off, 32);
  }
  float mean = sum * (1.0f / DD);
  float rstd = rsqrtf(sq * (1.0f / DD) - mean * mean + 1e-6f);
#pragma unroll
  for (int i = lane; i < DD; i += 32)
    xr[i] = (xp[i] - mean) * rstd * lnf_s[i] + lnf_b[i];
  __syncthreads();
  if (lane < NACTS) {
    float a = b_actor[lane];
    for (int i = 0; i < DD; ++i) a += xr[i] * W_actor[(size_t)i * NACTS + lane];
    logits[(size_t)row * NACTS + lane] = a;
  } else if (lane == NACTS) {
    float a = b_critic[0];
    for (int i = 0; i < DD; ++i) a += xr[i] * W_critic[i];
    val[row] = a;
  }
}

__global__ void sobs_kernel(const int* __restrict__ sin, int* __restrict__ sout) {
  int i = threadIdx.x;
  if (i < BB) sout[i] = sin[i] + TT;
}

// ---------------------------------------------------------------------------
// Host driver
// ---------------------------------------------------------------------------
extern "C" void kernel_launch(void* const* d_in, const int* in_sizes, int n_in,
                              void* d_out, int out_size, void* d_ws, size_t ws_size,
                              hipStream_t stream) {
  (void)in_sizes; (void)n_in; (void)out_size; (void)ws_size;
  const float* obs      = (const float*)d_in[0];
  const int*   act_p    = (const int*)  d_in[1];
  const float* rew_p    = (const float*)d_in[2];
  const int*   st_obs   = (const int*)  d_in[3];
  const float* st_blk   = (const float*)d_in[4];
  const float* W_obs    = (const float*)d_in[5];
  const float* b_obs    = (const float*)d_in[6];
  const float* E_act    = (const float*)d_in[7];
  const float* W_rew    = (const float*)d_in[8];
  const float* b_rew    = (const float*)d_in[9];
  const float* E_time   = (const float*)d_in[10];
  const float* ln1_s    = (const float*)d_in[11];
  const float* ln1_b    = (const float*)d_in[12];
  const float* Wqkv     = (const float*)d_in[13];
  const float* bqkv     = (const float*)d_in[14];
  const float* Wout     = (const float*)d_in[15];
  const float* bout     = (const float*)d_in[16];
  const float* ln2_s    = (const float*)d_in[17];
  const float* ln2_b    = (const float*)d_in[18];
  const float* W1       = (const float*)d_in[19];
  const float* b1       = (const float*)d_in[20];
  const float* W2       = (const float*)d_in[21];
  const float* b2       = (const float*)d_in[22];
  const float* lnf_s    = (const float*)d_in[23];
  const float* lnf_b    = (const float*)d_in[24];
  const float* W_actor  = (const float*)d_in[25];
  const float* b_actor  = (const float*)d_in[26];
  const float* W_critic = (const float*)d_in[27];
  const float* b_critic = (const float*)d_in[28];

  // ---- d_out partition: blocks[f32], sobs[i32], logits[f32], val[f32] ----
  float* out_blocks = (float*)d_out;                       // 64*4*8*64*64
  const size_t BLK_ELEMS = (size_t)BB * LL * HH * DH * DH; // 8388608
  int*   out_sobs   = (int*)(out_blocks + BLK_ELEMS);      // 64
  float* out_logits = out_blocks + BLK_ELEMS + BB;         // 64*512*16
  float* out_val    = out_logits + (size_t)BB * TT * NACTS;

  // ---- workspace layout ----
  char* ws = (char*)d_ws;
  size_t off = 0;
  auto take = [&](size_t bytes) -> char* {
    char* p = ws + off;
    off = (off + bytes + 255) & ~(size_t)255;
    return p;
  };
  _Float16* WT_obs = (_Float16*)take((size_t)DD * DOBS * 2);
  _Float16* WT_qkv = (_Float16*)take((size_t)LL * 3 * DD * DD * 2);
  _Float16* WT_out = (_Float16*)take((size_t)LL * DD * DD * 2);
  _Float16* WT_1   = (_Float16*)take((size_t)LL * DFF * DD * 2);
  _Float16* WT_2   = (_Float16*)take((size_t)LL * DD * DFF * 2);
  float*    xbuf   = (float*)   take((size_t)ROWS * DD * 4);
  _Float16* abuf   = (_Float16*)take((size_t)ROWS * DD * 2);
  _Float16* qkvbuf = (_Float16*)take((size_t)ROWS * 3 * DD * 2);
  _Float16* obuf   = (_Float16*)take((size_t)ROWS * DD * 2);
  _Float16* hbuf   = (_Float16*)take((size_t)ROWS * DFF * 2);

  // ---- weight conversion (f32 -> f16, transposed) ----
  cvt_t_kernel<<<256, 256, 0, stream>>>(W_obs, WT_obs, DOBS, DD);
  for (int l = 0; l < LL; ++l) {
    cvt_t_kernel<<<512, 256, 0, stream>>>(Wqkv + (size_t)l * DD * 3 * DD,
                                          WT_qkv + (size_t)l * 3 * DD * DD, DD, 3 * DD);
    cvt_t_kernel<<<256, 256, 0, stream>>>(Wout + (size_t)l * DD * DD,
                                          WT_out + (size_t)l * DD * DD, DD, DD);
    cvt_t_kernel<<<512, 256, 0, stream>>>(W1 + (size_t)l * DD * DFF,
                                          WT_1 + (size_t)l * DFF * DD, DD, DFF);
    cvt_t_kernel<<<512, 256, 0, stream>>>(W2 + (size_t)l * DFF * DD,
                                          WT_2 + (size_t)l * DD * DFF, DFF, DD);
  }

  // ---- embedding ----
  embed_kernel<<<ROWS, 256, 0, stream>>>(obs, act_p, rew_p, st_obs, W_obs, b_obs,
                                         E_act, W_rew, b_rew, E_time, xbuf);

  dim3 blk256(256);
  const int MB64 = ROWS / 64;                   // 512 M block-tiles
  for (int l = 0; l < LL; ++l) {
    // LN1 -> f16
    ln_kernel<<<ROWS / 8, blk256, 0, stream>>>(xbuf, abuf, ln1_s + l * DD, ln1_b + l * DD);
    // QKV GEMM: [32768,512] x [512,1536]
    gemm_kernel<0><<<dim3((3 * DD) / 128, MB64), blk256, 0, stream>>>(
        abuf, WT_qkv + (size_t)l * 3 * DD * DD, bqkv + (size_t)l * 3 * DD,
        qkvbuf, nullptr, 3 * DD, DD);
    // chunked linear attention + state carry (writes new_state_blocks)
    attn_kernel<<<dim3(HH, BB), 128, 0, stream>>>(qkvbuf, st_blk, out_blocks, obuf, l);
    // out projection + residual into x (f32)
    gemm_kernel<2><<<dim3(DD / 128, MB64), blk256, 0, stream>>>(
        obuf, WT_out + (size_t)l * DD * DD, bout + (size_t)l * DD,
        nullptr, xbuf, DD, DD);
    // LN2 -> f16
    ln_kernel<<<ROWS / 8, blk256, 0, stream>>>(xbuf, abuf, ln2_s + l * DD, ln2_b + l * DD);
    // FFN1 + gelu
    gemm_kernel<1><<<dim3(DFF / 128, MB64), blk256, 0, stream>>>(
        abuf, WT_1 + (size_t)l * DFF * DD, b1 + (size_t)l * DFF,
        hbuf, nullptr, DFF, DD);
    // FFN2 + residual into x
    gemm_kernel<2><<<dim3(DD / 128, MB64), blk256, 0, stream>>>(
        hbuf, WT_2 + (size_t)l * DD * DFF, b2 + (size_t)l * DD,
        nullptr, xbuf, DD, DFF);
  }

  // ---- final LN + heads ----
  head_kernel<<<ROWS, 32, 0, stream>>>(xbuf, lnf_s, lnf_b, W_actor, b_actor,
                                       W_critic, b_critic, out_logits, out_val);
  sobs_kernel<<<1, 64, 0, stream>>>(st_obs, out_sobs);
}
